// PPRGo_54674933678408
// MI455X (gfx1250) — matrix operand order repaired and checked by reading, not compiled, not run on hardware
//
#include <hip/hip_runtime.h>

// PPRGo forward: out[n, e] = sum_k (mask[n,k]/count[n]) * emb[nei[n,k], e]
// Memory-bound gather (~1.7 GB traffic; 51 MB table resides in 192 MB L2).
// Per-node 1x32 @ 32x64 reduction done with V_WMMA_F32_16X16X32_F16 via the
// broadcast-A trick (all 16 A rows = w, so every D row is the answer).
//
// Transpose staging: lane k gathers row nei[n,k] (16x b128). Row pairs are
// packed with v_cvt_pk_f16_f32, exchanged across lane pairs with one DPP
// quad_perm, and each output row dword extracted with a single v_perm_b32
// whose selector encodes the lane parity (hoisted constant). The tile is
// written with ds_store_2addr_b32 (compiler-merged) and read back as two
// contiguous ds_load_b128 per 16-wide e-chunk.

typedef __attribute__((ext_vector_type(16))) _Float16 v16h;
typedef __attribute__((ext_vector_type(8)))  _Float16 v8h;
typedef __attribute__((ext_vector_type(2)))  _Float16 h2;
typedef __attribute__((ext_vector_type(8)))  float    v8f;

#define TOPK 32
#define EMB  64
#define KHALVES 40            // halves per e-row in LDS (32 used + 8 pad)
#define WAVES_PER_BLOCK 8
#define THREADS (WAVES_PER_BLOCK * 32)

__global__ __launch_bounds__(THREADS) void pprgo_wmma_kernel(
    const float* __restrict__ emb,        // [N, EMB] f32
    const float* __restrict__ wei,        // [N, TOPK] f32
    const long long* __restrict__ nei,    // [N, TOPK] i64
    float* __restrict__ out,              // [N, EMB] f32
    int n_nodes)
{
    // Per-wave transposed tile: tile[e * KHALVES + k], f16.
    __shared__ __align__(16) _Float16 lds [WAVES_PER_BLOCK][EMB * KHALVES];
    __shared__ __align__(16) _Float16 wlds[WAVES_PER_BLOCK][TOPK];

    const int lane = threadIdx.x & 31;
    const int wib  = threadIdx.x >> 5;
    _Float16* tile = lds[wib];
    _Float16* wb   = wlds[wib];

    const int wave_id = blockIdx.x * WAVES_PER_BLOCK + wib;
    const int n_waves = gridDim.x * WAVES_PER_BLOCK;

    const bool hi   = (lane >= 16);
    const int  col  = lane & 15;
    const int  koff = hi ? 16 : 0;
    const int  par  = lane & 1;          // position within lane pair
    const int  cp   = lane & ~1;         // column-pair base (halves) in tile row
    // v_perm selector (per-lane constant): from (mine, part) extract
    //  even lane: row e   = (mine.lo, part.lo) -> bytes {4,5,0,1}
    //  odd  lane: row e+1 = (part.hi, mine.hi) -> bytes {2,3,6,7}
    const unsigned psel = par ? 0x07060302u : 0x01000504u;

    for (int node = wave_id; node < n_nodes; node += n_waves) {
        const long long base = (long long)node * TOPK;

        // Prefetch next iteration's index/weight cachelines.
        const int nxt = node + n_waves;
        if (nxt < n_nodes) {
            __builtin_prefetch(nei + (long long)nxt * TOPK + lane, 0, 0);
            __builtin_prefetch(wei + (long long)nxt * TOPK + lane, 0, 0);
        }

        // ---- weights: w_k = mask_k / (count + 1e-12), lane k owns k ----
        const float wv = wei[base + lane];
        const unsigned mbits = (unsigned)__ballot(wv != 0.0f);
        const float cnt = (float)__popc(mbits);
        const float w = (wv != 0.0f) ? (1.0f / (cnt + 1e-12f)) : 0.0f;

        // Stage w (f16) for the A-operand broadcast (2 contiguous 16B chunks).
        wb[lane] = (_Float16)w;                         // ds_store_b16

        // ---- gather: lane k streams row nei[n,k]; pair-pack via DPP ----
        const long long r = nei[base + lane];
        const float* row = emb + r * (long long)EMB;
        #pragma unroll
        for (int j = 0; j < EMB / 4; ++j) {             // e = 4j .. 4j+3
            const float4 v = *(const float4*)(row + j * 4);
            // Pack adjacent rows (same k): fuses to v_cvt_pk_f16_f32 (RTNE).
            h2 pa; pa[0] = (_Float16)v.x; pa[1] = (_Float16)v.y; // rows 4j,4j+1
            h2 pb; pb[0] = (_Float16)v.z; pb[1] = (_Float16)v.w; // rows 4j+2,4j+3
            const unsigned mine0 = __builtin_bit_cast(unsigned, pa);
            const unsigned mine1 = __builtin_bit_cast(unsigned, pb);
            const unsigned part0 = (unsigned)__builtin_amdgcn_update_dpp(
                0, (int)mine0, 0xB1 /*quad_perm(1,0,3,2)*/, 0xF, 0xF, true);
            const unsigned part1 = (unsigned)__builtin_amdgcn_update_dpp(
                0, (int)mine1, 0xB1, 0xF, 0xF, true);
            // Row dwords (col cp, cp+1); parity baked into the selector.
            const unsigned s0 = __builtin_amdgcn_perm(mine0, part0, psel);
            const unsigned s1 = __builtin_amdgcn_perm(mine1, part1, psel);
            const int e0 = j * 4 + par;   // even lane: rows 4j,4j+2; odd: +1
            *(h2*)(tile + e0 * KHALVES + cp)       = __builtin_bit_cast(h2, s0);
            *(h2*)(tile + (e0 + 2) * KHALVES + cp) = __builtin_bit_cast(h2, s1);
        }

        // ---- A operand: broadcast w rows (16-bit A 16x32 layout) ----
        // half h of lane L holds K = h + (h>=8 ? 8 : 0) + (L>=16 ? 8 : 0)
        const v8h a0 = *(const v8h*)(wb + koff / 2);        // K  0.. 7 / 8..15
        const v8h a1 = *(const v8h*)(wb + 16 + koff / 2);   // K 16..23 /24..31
        v16h a;
        #pragma unroll
        for (int h = 0; h < 8; ++h) { a[h] = a0[h]; a[h + 8] = a1[h]; }

        // ---- 4 e-chunks of 16: B = E_g[32 x 16] from LDS, one WMMA each ----
        v8f acc[4];
        #pragma unroll
        for (int c = 0; c < 4; ++c) {
            const _Float16* src = tile + (c * 16 + col) * KHALVES + koff / 2;
            const v8h b0 = *(const v8h*)(src);       // K 0..7 (lo lanes) / 8..15 (hi lanes)
            const v8h b1 = *(const v8h*)(src + 16);  // K 16..23 (lo lanes) / 24..31 (hi lanes)
            v16h b;
            #pragma unroll
            for (int h = 0; h < 8; ++h) { b[h] = b0[h]; b[h + 8] = b1[h]; }
            v8f z = {};
            acc[c] = __builtin_amdgcn_wmma_f32_16x16x32_f16(
                false, a, false, b, (short)0, z, false, false);
        }

        // ---- store: all D rows identical (broadcast A), so lanes 16..31
        // (VGPR0 = row M=8) carry the same values as row M=0. Merge chunk
        // pairs into two full-wave b32 stores.
        const float o01 = hi ? acc[1][0] : acc[0][0];   // e = lane
        const float o23 = hi ? acc[3][0] : acc[2][0];   // e = 32 + lane
        float* orow = out + (long long)node * EMB;
        orow[lane]      = o01;
        orow[lane + 32] = o23;
    }
}

extern "C" void kernel_launch(void* const* d_in, const int* in_sizes, int n_in,
                              void* d_out, int out_size, void* d_ws, size_t ws_size,
                              hipStream_t stream) {
    const float*     emb = (const float*)d_in[0];      // [N, 64] f32
    const float*     wei = (const float*)d_in[1];      // [N, 32] f32
    const long long* nei = (const long long*)d_in[2];  // [N, 32] i64
    float* out = (float*)d_out;                        // [N, 1, 64] f32 flat

    const int n_nodes = in_sizes[1] / TOPK;

    int blocks = (n_nodes + WAVES_PER_BLOCK - 1) / WAVES_PER_BLOCK;
    if (blocks > 2048) blocks = 2048;   // 16K waves, grid-stride over nodes
    if (blocks < 1) blocks = 1;

    pprgo_wmma_kernel<<<blocks, THREADS, 0, stream>>>(emb, wei, nei, out, n_nodes);
}
